// BiLSTM_CRF_41515153883149
// MI455X (gfx1250) — matrix operand order repaired and checked by reading, not compiled, not run on hardware
//
#include <hip/hip_runtime.h>
#include <hip/hip_bf16.h>

// ---------------------------------------------------------------------------
// CRF forward (BiLSTM-CRF NLL) for MI455X / gfx1250.
//
// Reformulation: logsumexp DP  ->  f16 WMMA matvec against exp(transitions)
//   alpha_new[b,i] = m_b + log( sum_j expT[i,j] * v[b,j] ) + feat_t[b,i]
// with v[b,j] = exp(alpha[b,j] - m_b) in [0,1] (f16), m_b a running f32 offset.
// Each step: [16b x 128j] x [128j x 128i] == 8 waves x 4 v_wmma_f32_16x16x32_f16.
// Latency-bound (512 sequential steps); sched_barriers keep all 8 ds_load_b128
// ahead of the 4 WMMAs so LDS latency is paid once per step, not 4x.
// ---------------------------------------------------------------------------

typedef __attribute__((ext_vector_type(16))) _Float16 v16h;
typedef __attribute__((ext_vector_type(8)))  _Float16 v8h;
typedef __attribute__((ext_vector_type(8)))  float    v8f;

#define NTAG 128
#define TT   512
#define START_TAG 126
#define STOP_TAG  127
#define VPAD 8   // pad vmat rows: row stride 136 halves = 272B -> bank-conflict-free A loads

// ---- prep: expT[j*128+i] = exp(transitions[i*128+j])  (B-operand, transposed, f16)
__global__ void mexp_kernel(const float* __restrict__ trans,
                            _Float16* __restrict__ mexpT) {
    int idx = blockIdx.x * blockDim.x + threadIdx.x;   // 16384 threads
    int i = idx >> 7, j = idx & 127;
    mexpT[j * NTAG + i] = (_Float16)__expf(trans[i * NTAG + j]);
}

// ---- gold path score: exact f32 gather/sum, one block per sequence
__global__ void gold_kernel(const float* __restrict__ feats,
                            const float* __restrict__ trans,
                            const int*   __restrict__ tags,
                            float* __restrict__ gold) {
    int b = blockIdx.x, tid = threadIdx.x;             // 128 threads
    float s = 0.0f;
    for (int t = tid; t < TT; t += 128) {
        int tg   = tags[b * TT + t];
        int prev = (t == 0) ? START_TAG : tags[b * TT + t - 1];
        s += feats[((size_t)b * TT + t) * NTAG + tg] + trans[tg * NTAG + prev];
    }
    if (tid == 0) s += trans[STOP_TAG * NTAG + tags[b * TT + TT - 1]];
    __shared__ float red[128];
    red[tid] = s; __syncthreads();
    for (int off = 64; off > 0; off >>= 1) {
        if (tid < off) red[tid] += red[tid + off];
        __syncthreads();
    }
    if (tid == 0) gold[b] = red[0];
}

// ---- main forward scan: 32 blocks x 256 threads (8 waves), 16 batches/block
__launch_bounds__(256)
__global__ void crf_forward_kernel(const float* __restrict__ feats,
                                   const _Float16* __restrict__ mexpT,
                                   const float* __restrict__ gold,
                                   float* __restrict__ out) {
    __shared__ _Float16 vmat[16][NTAG + VPAD]; // v = exp(alpha - m), f16
    __shared__ float    zbuf[16][NTAG];        // z = log(u) + feat  (pre-renorm)
    __shared__ float    mBuf[16];              // running log-offset per batch

    const int tid  = threadIdx.x;
    const int w    = tid >> 5;          // wave id: owns i in [16w, 16w+16)
    const int lane = tid & 31;
    const int n    = lane & 15;         // WMMA N / A-row index
    const int hi   = lane >> 4;         // lane half select
    const int b0   = blockIdx.x * 16;   // batch tile base

    // ---- init state: alpha = -inf except START=0  =>  v = one-hot(START), m = 0
    {
        int b = tid >> 4, seg = tid & 15;
        #pragma unroll
        for (int k = 0; k < 8; ++k) {
            int j = seg * 8 + k;
            vmat[b][j] = (j == START_TAG) ? (_Float16)1.0f : (_Float16)0.0f;
        }
        if (seg == 0) mBuf[b] = 0.0f;
    }

    // ---- constant B tiles (held in VGPRs for the whole scan)
    // CDNA5 f16 B 32x16 layout: lane (n,hi), element e  ->  K = 8*hi + e (e<8), 16 + 8*hi + (e-8) (e>=8), N = n
    v16h Bt[4];
    #pragma unroll
    for (int c = 0; c < 4; ++c) {
        #pragma unroll
        for (int e = 0; e < 16; ++e) {
            int j = 32 * c + 8 * hi + e + (e >= 8 ? 8 : 0); // K dim = source tag j
            Bt[c][e] = mexpT[j * NTAG + (16 * w + n)]; // N dim = dest tag i
        }
    }

    // ---- feat prefetch registers in WMMA D layout: reg r -> batch r+8*hi, tag 16w+n
    float fcur[8], fnext[8];
    #pragma unroll
    for (int r = 0; r < 8; ++r)
        fcur[r] = feats[((size_t)(b0 + r + 8 * hi) * TT + 0) * NTAG + 16 * w + n];

    __syncthreads();

    for (int t = 0; t < TT; ++t) {
        // ---- Phase A: build A tiles from LDS v (CDNA5 16-bit A layout:
        // lane row m=n; halves 0-7 -> K = 8*hi + e, halves 8-15 -> K = 16 + 8*hi + (e-8))
        v16h A[4];
        #pragma unroll
        for (int c = 0; c < 4; ++c) {
            v8h lo = *(const v8h*)&vmat[n][32 * c + 8 * hi];
            v8h hh = *(const v8h*)&vmat[n][32 * c + 16 + 8 * hi];
            A[c] = __builtin_shufflevector(lo, hh, 0,1,2,3,4,5,6,7,8,9,10,11,12,13,14,15);
        }

        // prefetch next step's feats while WMMAs run
        int tn = (t + 1 < TT) ? (t + 1) : t;
        #pragma unroll
        for (int r = 0; r < 8; ++r)
            fnext[r] = feats[((size_t)(b0 + r + 8 * hi) * TT + tn) * NTAG + 16 * w + n];

        // keep all 8 ds_load_b128 (+ the global clause) issued BEFORE the WMMA
        // chain: one s_wait_dscnt instead of four LDS round-trips per step
        __builtin_amdgcn_sched_barrier(0);

        // u[b, i-tile] = sum over 128 j, as 4 x K=32 WMMA, f32 accumulate
        v8f acc = {};
        #pragma unroll
        for (int c = 0; c < 4; ++c)
            acc = __builtin_amdgcn_wmma_f32_16x16x32_f16(
                false, A[c], false, Bt[c], (short)0, acc, false, false);

        // z = log(u) + feat  (D layout: reg r -> batch r+8*hi, tag 16w+n)
        #pragma unroll
        for (int r = 0; r < 8; ++r)
            zbuf[r + 8 * hi][16 * w + n] = __logf(acc[r]) + fcur[r];
        __syncthreads();

        // ---- Phase B: per-batch renormalize: dm = max_i z, v = exp(z-dm), m += dm
        {
            int b = tid >> 4, seg = tid & 15;   // 16 threads per batch x 8 values
            float z[8];
            #pragma unroll
            for (int k = 0; k < 8; ++k) z[k] = zbuf[b][seg * 8 + k];
            float mx = z[0];
            #pragma unroll
            for (int k = 1; k < 8; ++k) mx = fmaxf(mx, z[k]);
            #pragma unroll
            for (int s = 1; s < 16; s <<= 1)      // reduce across the 16-lane group
                mx = fmaxf(mx, __shfl_xor(mx, s, 32));
            v8h vh;
            #pragma unroll
            for (int k = 0; k < 8; ++k) vh[k] = (_Float16)__expf(z[k] - mx);
            *(v8h*)&vmat[b][seg * 8] = vh;
            if (seg == 0) mBuf[b] += mx;
        }
        __syncthreads();

        #pragma unroll
        for (int r = 0; r < 8; ++r) fcur[r] = fnext[r];
    }

    // ---- termination: forward = m + log( sum_j exp(trans[STOP,j]) * v[b,j] )
    // Only wave 7's tile contains i = STOP (= 127 -> n == 15).
    if (w == 7) {
        v16h A[4];
        #pragma unroll
        for (int c = 0; c < 4; ++c) {
            v8h lo = *(const v8h*)&vmat[n][32 * c + 8 * hi];
            v8h hh = *(const v8h*)&vmat[n][32 * c + 16 + 8 * hi];
            A[c] = __builtin_shufflevector(lo, hh, 0,1,2,3,4,5,6,7,8,9,10,11,12,13,14,15);
        }
        __builtin_amdgcn_sched_barrier(0);
        v8f acc = {};
        #pragma unroll
        for (int c = 0; c < 4; ++c)
            acc = __builtin_amdgcn_wmma_f32_16x16x32_f16(
                false, A[c], false, Bt[c], (short)0, acc, false, false);
        if (n == 15) {                        // lanes 15 & 31 hold column i = STOP
            #pragma unroll
            for (int r = 0; r < 8; ++r) {
                int bl = r + 8 * hi;
                out[b0 + bl] = mBuf[bl] + __logf(acc[r]) - gold[b0 + bl];
            }
        }
    }
}

extern "C" void kernel_launch(void* const* d_in, const int* in_sizes, int n_in,
                              void* d_out, int out_size, void* d_ws, size_t ws_size,
                              hipStream_t stream) {
    const float* feats = (const float*)d_in[0];   // [512, 512, 128] f32
    const float* trans = (const float*)d_in[1];   // [128, 128] f32
    const int*   tags  = (const int*)d_in[2];     // [512, 512] i32
    float* out = (float*)d_out;                   // [512] f32

    float*    gold  = (float*)d_ws;                                   // 512 f32
    _Float16* mexpT = (_Float16*)((char*)d_ws + 512 * sizeof(float)); // 128x128 f16

    mexp_kernel<<<64, 256, 0, stream>>>(trans, mexpT);
    gold_kernel<<<512, 128, 0, stream>>>(feats, trans, tags, gold);
    crf_forward_kernel<<<32, 256, 0, stream>>>(feats, mexpT, gold, out);
}
